// MEGADecoder_35390530519894
// MI455X (gfx1250) — compile-verified, matrix operand order checked by reading, not run on hardware
//
#include <hip/hip_runtime.h>
#include <hip/hip_fp16.h>
#include <stdint.h>

typedef __attribute__((ext_vector_type(16))) _Float16 v16h;
typedef __attribute__((ext_vector_type(8)))  _Float16 v8h;
typedef __attribute__((ext_vector_type(8)))  float    v8f;
typedef __attribute__((ext_vector_type(4)))  unsigned su4;
typedef __attribute__((ext_vector_type(8)))  unsigned su8;

#define TPB 256
#define CEILDIV(a,b) (((a)+(b)-1)/(b))

enum { ACT_NONE = 0, ACT_TANH = 1, ACT_SILU = 2, ACT_SIG = 3 };

// ---------------------------------------------------------------------------
// TDM descriptor-driven tile load: copy a 128-row x 64-col fp16 tile whose
// rows stride Kdim elements in global memory into LDS at lds_off, inserting
// 16B of padding after every 128B row (pad_interval=32 DW, pad_amount=4 DW)
// so fragment ds_load_b128s are bank-conflict free. Tracked by TENSORcnt.
// ---------------------------------------------------------------------------
__device__ __forceinline__ void tdm_load_tile_f16(const _Float16* gtile,
                                                  unsigned lds_off,
                                                  int Kdim, int Rows)
{
    unsigned long long ga = (unsigned long long)(uintptr_t)gtile;

    su4 g0;
    g0[0] = 1u;                                            // count=1 (valid), user mode
    g0[1] = lds_off;                                       // lds_addr
    g0[2] = (unsigned)(ga & 0xFFFFFFFFu);                  // global_addr[31:0]
    g0[3] = (unsigned)((ga >> 32) & 0x1FFFFFFu)            // global_addr[56:32]
          | (2u << 30);                                    // type=2 ("image")

    su8 g1;
    g1[0] = (1u << 16)                                     // data_size = 2 bytes
          | (1u << 20)                                     // pad_enable
          | (4u << 22)                                     // pad_interval: 32 DWORDs (128B)
          | (3u << 25);                                    // pad_amount: 4 DWORDs (16B)
    g1[1] = ((unsigned)Kdim & 0xFFFFu) << 16;              // tensor_dim0[15:0] (bits 63:48)
    g1[2] = (((unsigned)Kdim >> 16) & 0xFFFFu)             // tensor_dim0[31:16]
          | (((unsigned)Rows & 0xFFFFu) << 16);            // tensor_dim1[15:0]
    g1[3] = (((unsigned)Rows >> 16) & 0xFFFFu)             // tensor_dim1[31:16]
          | (64u << 16);                                   // tile_dim0 = 64 elements
    g1[4] = 128u;                                          // tile_dim1 = 128 rows, tile_dim2=0
    g1[5] = (unsigned)Kdim;                                // tensor_dim0_stride[31:0]
    g1[6] = 0u;                                            // stride[47:32], dim1_stride[15:0]
    g1[7] = 0u;

    asm volatile("tensor_load_to_lds %0, %1" :: "s"(g0), "s"(g1) : "memory");
}

// ---------------------------------------------------------------------------
// Tiled WMMA GEMM:  C[M,N] = act( scale * (A[M,K] . B[N,K]^T) + bias[N] )
// A, B fp16 row-major, K contiguous. 128x128 block tile, BK=64 staged tiles.
// Double-buffered LDS filled by the Tensor Data Mover (one descriptor per
// tile, issued by wave 0), overlapping tile k+1 DMA with tile k WMMA compute.
// 8 waves/block; each wave owns a 32x64 sub-tile = 2x4 WMMA 16x16 tiles.
// ---------------------------------------------------------------------------
template<int ACT, bool OUTF, bool OUTH>
__global__ __launch_bounds__(256)
void mega_gemm_nt_wmma(const _Float16* __restrict__ A,
                       const _Float16* __restrict__ B,
                       const float* __restrict__ bias,
                       float scale,
                       float* __restrict__ Cf,
                       _Float16* __restrict__ Ch,
                       int M, int N, int K)
{
    constexpr int BM = 128, BN = 128, BK = 64;
    constexpr int LDP = BK + 8;          // 144B row stride (16B-aligned, pad via TDM)
    __shared__ _Float16 As[2][BM][LDP];  // 36 KB
    __shared__ _Float16 Bs[2][BN][LDP];  // 36 KB

    const int tid  = threadIdx.x;
    const int lane = tid & 31;
    const int wave = tid >> 5;
    const int wr   = wave >> 1;        // 0..3  (wave row)
    const int wc   = wave & 1;         // 0..1  (wave col)
    const int mBase = wr * 32;
    const int nBase = wc * 64;
    const int blockM = blockIdx.y * BM;
    const int blockN = blockIdx.x * BN;

    const int fr = lane & 15;          // fragment row (M for A, N for B)
    const int kq = lane >> 4;          // K-half selector per ISA layout

    v8f acc[2][4] = {};

    // Pipeline prologue: DMA tile 0 into buffer 0 (wave 0 drives the TDM).
    if (wave == 0) {
        tdm_load_tile_f16(&A[(size_t)blockM * K],
                          (unsigned)(uintptr_t)&As[0][0][0], K, M);
        tdm_load_tile_f16(&B[(size_t)blockN * K],
                          (unsigned)(uintptr_t)&Bs[0][0][0], K, N);
    }
    __builtin_amdgcn_s_wait_tensorcnt(0);
    __syncthreads();

    const int nIter = K / BK;
    for (int i = 0; i < nIter; ++i) {
        const int buf = i & 1;

        // Kick off the next tile's DMA before computing this one.
        if (i + 1 < nIter && wave == 0) {
            int k0 = (i + 1) * BK;
            tdm_load_tile_f16(&A[(size_t)blockM * K + k0],
                              (unsigned)(uintptr_t)&As[buf ^ 1][0][0], K, M);
            tdm_load_tile_f16(&B[(size_t)blockN * K + k0],
                              (unsigned)(uintptr_t)&Bs[buf ^ 1][0][0], K, N);
        }

        // Compute: 2 WMMA K-steps over the staged BK=64 tile.
        #pragma unroll
        for (int ks = 0; ks < BK; ks += 32) {
            v16h af[2], bf[4];
            #pragma unroll
            for (int ii = 0; ii < 2; ++ii) {
                const _Float16* p = &As[buf][mBase + ii * 16 + fr][ks + kq * 8];
                v8h lo = *(const v8h*)p;
                v8h hi = *(const v8h*)(p + 16);
                #pragma unroll
                for (int t = 0; t < 8; ++t) { af[ii][t] = lo[t]; af[ii][t + 8] = hi[t]; }
            }
            #pragma unroll
            for (int j = 0; j < 4; ++j) {
                const _Float16* p = &Bs[buf][nBase + j * 16 + fr][ks + kq * 8];
                v8h lo = *(const v8h*)p;
                v8h hi = *(const v8h*)(p + 16);
                #pragma unroll
                for (int t = 0; t < 8; ++t) { bf[j][t] = lo[t]; bf[j][t + 8] = hi[t]; }
            }
            #pragma unroll
            for (int ii = 0; ii < 2; ++ii)
                #pragma unroll
                for (int j = 0; j < 4; ++j)
                    acc[ii][j] = __builtin_amdgcn_wmma_f32_16x16x32_f16(
                        false, af[ii], false, bf[j], (short)0, acc[ii][j], false, false);
        }

        // Stage boundary: the next tile's DMA has landed; make it visible.
        __builtin_amdgcn_s_wait_tensorcnt(0);
        __syncthreads();
    }

    // Epilogue. C/D VGPR layout: VGPR v -> (M = v + 8*(lane>=16), N = lane&15).
    const int n16 = lane & 15;
    const int mh  = lane >> 4;
    #pragma unroll
    for (int ii = 0; ii < 2; ++ii) {
        #pragma unroll
        for (int j = 0; j < 4; ++j) {
            int col = blockN + nBase + j * 16 + n16;
            float bvv = bias ? bias[col] : 0.0f;
            #pragma unroll
            for (int v = 0; v < 8; ++v) {
                int row = blockM + mBase + ii * 16 + v + 8 * mh;
                float x = acc[ii][j][v] * scale + bvv;
                if constexpr (ACT == ACT_TANH) x = tanhf(x);
                else if constexpr (ACT == ACT_SILU) x = x / (1.0f + __expf(-x));
                else if constexpr (ACT == ACT_SIG)  x = 1.0f / (1.0f + __expf(-x));
                size_t o = (size_t)row * N + col;
                if constexpr (OUTF) Cf[o] = x;
                if constexpr (OUTH) Ch[o] = (_Float16)x;
            }
        }
    }
}

// ---------------------------------------------------------------------------
// Elementwise / helper kernels
// ---------------------------------------------------------------------------
__global__ __launch_bounds__(256)
void mega_f32_to_f16(const float* __restrict__ in, _Float16* __restrict__ out, int n)
{
    int i = blockIdx.x * blockDim.x + threadIdx.x;
    if (i < n) out[i] = (_Float16)in[i];
}

__global__ __launch_bounds__(256)
void mega_build_concat(const float* __restrict__ R, _Float16* __restrict__ cat,
                       int S, int D)
{
    int i = blockIdx.x * blockDim.x + threadIdx.x;
    int n = S * 2 * D;
    if (i >= n) return;
    int m = i / (2 * D);
    int c = i % (2 * D);
    float v;
    if (c < D) v = (m == 0) ? 0.0f : R[(size_t)(m - 1) * D + c];   // r_prev
    else       v = R[(size_t)m * D + (c - D)];                     // r_t
    cat[i] = (_Float16)v;
}

__global__ __launch_bounds__(256)
void mega_ema_combine(const float* __restrict__ R,
                      const float* __restrict__ alpha,
                      const float* __restrict__ delta,
                      float* __restrict__ Ef, _Float16* __restrict__ Eh,
                      int S, int D)
{
    int i = blockIdx.x * blockDim.x + threadIdx.x;
    if (i >= S * D) return;
    int m = i / D;
    float rt = R[i];
    float rp = (m == 0) ? 0.0f : R[i - D];
    float a  = alpha[i];
    float dl = delta[i];
    float e  = a * rt + (1.0f - a) * dl * rp;
    Ef[i] = e;
    Eh[i] = (_Float16)e;
}

__global__ __launch_bounds__(256)
void mega_gate_f16(const float* __restrict__ f, const float* __restrict__ zat,
                   _Float16* __restrict__ out, int n)
{
    int i = blockIdx.x * blockDim.x + threadIdx.x;
    if (i < n) out[i] = (_Float16)(f[i] * zat[i]);
}

__global__ __launch_bounds__(256)
void mega_tanh_add(const float* __restrict__ a, const float* __restrict__ b,
                   float* __restrict__ out, int n)
{
    int i = blockIdx.x * blockDim.x + threadIdx.x;
    if (i < n) out[i] = tanhf(a[i] + b[i]);
}

// fp16 transpose: in[R x C] -> out[C x R]
__global__ __launch_bounds__(256)
void mega_transpose_f16(const _Float16* __restrict__ in, _Float16* __restrict__ out,
                        int R, int C)
{
    __shared__ _Float16 t[32][33];
    int bx = blockIdx.x * 32;   // col origin
    int by = blockIdx.y * 32;   // row origin
    int x = bx + threadIdx.x;
    for (int i = threadIdx.y; i < 32; i += 8)
        t[i][threadIdx.x] = in[(size_t)(by + i) * C + x];
    __syncthreads();
    int ox = by + threadIdx.x;
    for (int i = threadIdx.y; i < 32; i += 8)
        out[(size_t)(bx + i) * R + ox] = t[threadIdx.x][i];
}

// row softmax over N fp32 scores -> fp16 weights
__global__ __launch_bounds__(256)
void mega_softmax_rows(const float* __restrict__ S, _Float16* __restrict__ W, int N)
{
    __shared__ float red[TPB];
    const int row = blockIdx.x;
    const int tid = threadIdx.x;
    const float* s = S + (size_t)row * N;

    float m = -3.4e38f;
    for (int c = tid; c < N; c += TPB) m = fmaxf(m, s[c]);
    red[tid] = m; __syncthreads();
    for (int st = TPB / 2; st > 0; st >>= 1) {
        if (tid < st) red[tid] = fmaxf(red[tid], red[tid + st]);
        __syncthreads();
    }
    m = red[0]; __syncthreads();

    float sum = 0.0f;
    for (int c = tid; c < N; c += TPB) sum += __expf(s[c] - m);
    red[tid] = sum; __syncthreads();
    for (int st = TPB / 2; st > 0; st >>= 1) {
        if (tid < st) red[tid] += red[tid + st];
        __syncthreads();
    }
    float inv = 1.0f / red[0];

    for (int c = tid; c < N; c += TPB)
        W[(size_t)row * N + c] = (_Float16)(__expf(s[c] - m) * inv);
}

// per-row dot:  out[row] = tanh( X[row,:] . w + b[0] )
__global__ __launch_bounds__(256)
void mega_rowdot_tanh(const float* __restrict__ X, const float* __restrict__ w,
                      const float* __restrict__ b, float* __restrict__ out, int K)
{
    __shared__ float red[TPB];
    const int row = blockIdx.x;
    const int tid = threadIdx.x;
    float s = 0.0f;
    for (int k = tid; k < K; k += TPB) s += X[(size_t)row * K + k] * w[k];
    red[tid] = s; __syncthreads();
    for (int st = TPB / 2; st > 0; st >>= 1) {
        if (tid < st) red[tid] += red[tid + st];
        __syncthreads();
    }
    if (tid == 0) out[row] = tanhf(red[0] + b[0]);
}

// final:  p_hat[row] = sigmoid( sum_g (i*Ztp + (1-i)*Remap) * w_final[g] )
__global__ __launch_bounds__(256)
void mega_final(const float* __restrict__ iv, const float* __restrict__ Ztp,
                const float* __restrict__ Remap, const float* __restrict__ wfin,
                float* __restrict__ out, int K)
{
    __shared__ float red[TPB];
    const int row = blockIdx.x;
    const int tid = threadIdx.x;
    float g = iv[row];
    float s = 0.0f;
    for (int k = tid; k < K; k += TPB) {
        float zf = g * Ztp[(size_t)row * K + k] + (1.0f - g) * Remap[(size_t)row * K + k];
        s += zf * wfin[k];
    }
    red[tid] = s; __syncthreads();
    for (int st = TPB / 2; st > 0; st >>= 1) {
        if (tid < st) red[tid] += red[tid + st];
        __syncthreads();
    }
    if (tid == 0) out[row] = 1.0f / (1.0f + __expf(-red[0]));
}

// ---------------------------------------------------------------------------
// Host orchestration
// ---------------------------------------------------------------------------
extern "C" void kernel_launch(void* const* d_in, const int* in_sizes, int n_in,
                              void* d_out, int out_size, void* d_ws, size_t ws_size,
                              hipStream_t stream)
{
    (void)in_sizes; (void)n_in; (void)out_size; (void)ws_size;

    const int S = 4096, D = 1024;            // SEQ, R/Z/AT/G dims (all 1024)

    const float* R      = (const float*)d_in[0];
    const float* Walpha = (const float*)d_in[1];
    const float* balpha = (const float*)d_in[2];
    const float* Wdelta = (const float*)d_in[3];
    const float* bdelta = (const float*)d_in[4];
    const float* Wq     = (const float*)d_in[5];
    const float* bq     = (const float*)d_in[6];
    const float* Wk     = (const float*)d_in[7];
    const float* bk     = (const float*)d_in[8];
    const float* Wv     = (const float*)d_in[9];
    const float* bv     = (const float*)d_in[10];
    const float* Wz     = (const float*)d_in[11];
    const float* bz     = (const float*)d_in[12];
    const float* Wf     = (const float*)d_in[13];
    const float* bfb    = (const float*)d_in[14];
    const float* Wema   = (const float*)d_in[15];
    const float* bema   = (const float*)d_in[16];
    const float* Wzat   = (const float*)d_in[17];
    const float* bzat   = (const float*)d_in[18];
    const float* Wi     = (const float*)d_in[19];
    const float* bi     = (const float*)d_in[20];
    const float* Wfin   = (const float*)d_in[21];
    float* out = (float*)d_out;

    char* base = (char*)d_ws;
    size_t off = 0;
    auto alloc = [&](size_t bytes) -> void* {
        off = (off + 255) & ~(size_t)255;
        void* p = base + off;
        off += bytes;
        return p;
    };

    const size_t SD  = (size_t)S * D;         // 4M
    const size_t SS  = (size_t)S * S;         // 16M
    const size_t S2D = (size_t)S * 2 * D;     // 8M

    _Float16* cat_h    = (_Float16*)alloc(S2D * 2);
    _Float16* Walpha_h = (_Float16*)alloc((size_t)D * 2 * D * 2);
    _Float16* Wdelta_h = (_Float16*)alloc((size_t)D * 2 * D * 2);
    _Float16* Wz_h     = (_Float16*)alloc((size_t)D * D * 2);
    _Float16* Wq_h     = (_Float16*)alloc((size_t)D * D * 2);
    _Float16* Wk_h     = (_Float16*)alloc((size_t)D * D * 2);
    _Float16* Wv_h     = (_Float16*)alloc((size_t)D * D * 2);
    _Float16* Wf_h     = (_Float16*)alloc((size_t)D * D * 2);
    _Float16* Wema_h   = (_Float16*)alloc((size_t)D * D * 2);
    _Float16* Wzat_h   = (_Float16*)alloc((size_t)D * D * 2);

    float*    alpha_f  = (float*)alloc(SD * 4);
    float*    delta_f  = (float*)alloc(SD * 4);
    float*    ema_f    = (float*)alloc(SD * 4);
    _Float16* ema_h    = (_Float16*)alloc(SD * 2);
    _Float16* Z_h      = (_Float16*)alloc(SD * 2);
    _Float16* Q_h      = (_Float16*)alloc(SD * 2);
    _Float16* K_h      = (_Float16*)alloc(SD * 2);
    _Float16* V_h      = (_Float16*)alloc(SD * 2);
    _Float16* Vt_h     = (_Float16*)alloc(SD * 2);
    float*    scores_f = (float*)alloc(SS * 4);
    _Float16* att_h    = (_Float16*)alloc(SS * 2);
    float*    Zat_f    = (float*)alloc(SD * 4);
    float*    Remap_f  = (float*)alloc(SD * 4);
    _Float16* Remap_h  = (_Float16*)alloc(SD * 2);
    float*    fgate_f  = (float*)alloc(SD * 4);
    _Float16* Zatp_h   = (_Float16*)alloc(SD * 2);
    float*    Zemaf_f  = (float*)alloc(SD * 4);
    float*    Ztp_f    = (float*)alloc(SD * 4);
    float*    i_f      = (float*)alloc((size_t)S * 4);

    // --- stage 0: fp16 conversions + concat ---------------------------------
    mega_build_concat<<<CEILDIV((int)S2D, TPB), TPB, 0, stream>>>(R, cat_h, S, D);
    mega_f32_to_f16<<<CEILDIV(D * 2 * D, TPB), TPB, 0, stream>>>(Walpha, Walpha_h, D * 2 * D);
    mega_f32_to_f16<<<CEILDIV(D * 2 * D, TPB), TPB, 0, stream>>>(Wdelta, Wdelta_h, D * 2 * D);
    mega_f32_to_f16<<<CEILDIV(D * D, TPB), TPB, 0, stream>>>(Wz,   Wz_h,   D * D);
    mega_f32_to_f16<<<CEILDIV(D * D, TPB), TPB, 0, stream>>>(Wq,   Wq_h,   D * D);
    mega_f32_to_f16<<<CEILDIV(D * D, TPB), TPB, 0, stream>>>(Wk,   Wk_h,   D * D);
    mega_f32_to_f16<<<CEILDIV(D * D, TPB), TPB, 0, stream>>>(Wv,   Wv_h,   D * D);
    mega_f32_to_f16<<<CEILDIV(D * D, TPB), TPB, 0, stream>>>(Wf,   Wf_h,   D * D);
    mega_f32_to_f16<<<CEILDIV(D * D, TPB), TPB, 0, stream>>>(Wema, Wema_h, D * D);
    mega_f32_to_f16<<<CEILDIV(D * D, TPB), TPB, 0, stream>>>(Wzat, Wzat_h, D * D);

    dim3 blk(TPB);
    dim3 gDD(D / 128, S / 128);    // N=1024 GEMMs: 8 x 32 blocks
    dim3 gSS(S / 128, S / 128);    // N=4096 GEMMs: 32 x 32 blocks

    // --- EMA gates ----------------------------------------------------------
    mega_gemm_nt_wmma<ACT_TANH, true, false><<<gDD, blk, 0, stream>>>(
        cat_h, Walpha_h, balpha, 1.0f, alpha_f, nullptr, S, D, 2 * D);
    mega_gemm_nt_wmma<ACT_TANH, true, false><<<gDD, blk, 0, stream>>>(
        cat_h, Wdelta_h, bdelta, 1.0f, delta_f, nullptr, S, D, 2 * D);
    mega_ema_combine<<<CEILDIV((int)SD, TPB), TPB, 0, stream>>>(
        R, alpha_f, delta_f, ema_f, ema_h, S, D);

    // --- Z, Q, K, V ---------------------------------------------------------
    mega_gemm_nt_wmma<ACT_SILU, false, true><<<gDD, blk, 0, stream>>>(
        ema_h, Wz_h, bz, 1.0f, nullptr, Z_h, S, D, D);
    mega_gemm_nt_wmma<ACT_NONE, false, true><<<gDD, blk, 0, stream>>>(
        Z_h, Wq_h, bq, 1.0f, nullptr, Q_h, S, D, D);
    mega_gemm_nt_wmma<ACT_NONE, false, true><<<gDD, blk, 0, stream>>>(
        Z_h, Wk_h, bk, 1.0f, nullptr, K_h, S, D, D);
    mega_gemm_nt_wmma<ACT_NONE, false, true><<<gDD, blk, 0, stream>>>(
        Z_h, Wv_h, bv, 1.0f, nullptr, V_h, S, D, D);

    // --- attention ----------------------------------------------------------
    mega_gemm_nt_wmma<ACT_NONE, true, false><<<gSS, blk, 0, stream>>>(
        Q_h, K_h, nullptr, 0.03125f /* 1/sqrt(1024) */, scores_f, nullptr, S, S, D);
    mega_softmax_rows<<<S, TPB, 0, stream>>>(scores_f, att_h, S);
    {
        dim3 tb(32, 8), tg(D / 32, S / 32);
        mega_transpose_f16<<<tg, tb, 0, stream>>>(V_h, Vt_h, S, D);
    }
    mega_gemm_nt_wmma<ACT_NONE, true, false><<<gDD, blk, 0, stream>>>(
        att_h, Vt_h, nullptr, 1.0f, Zat_f, nullptr, S, D, S);

    // --- gating path --------------------------------------------------------
    mega_gemm_nt_wmma<ACT_NONE, true, true><<<gDD, blk, 0, stream>>>(
        ema_h, Wema_h, bema, 1.0f, Remap_f, Remap_h, S, D, D);
    mega_gemm_nt_wmma<ACT_SIG, true, false><<<gDD, blk, 0, stream>>>(
        Remap_h, Wf_h, bfb, 1.0f, fgate_f, nullptr, S, D, D);
    mega_gate_f16<<<CEILDIV((int)SD, TPB), TPB, 0, stream>>>(
        fgate_f, Zat_f, Zatp_h, (int)SD);
    mega_gemm_nt_wmma<ACT_NONE, true, false><<<gDD, blk, 0, stream>>>(
        Zatp_h, Wzat_h, bzat, 1.0f, Zemaf_f, nullptr, S, D, D);
    mega_tanh_add<<<CEILDIV((int)SD, TPB), TPB, 0, stream>>>(
        Remap_f, Zemaf_f, Ztp_f, (int)SD);

    // --- scalar gate i and final mix ---------------------------------------
    mega_rowdot_tanh<<<S, TPB, 0, stream>>>(ema_f, Wi, bi, i_f, D);
    mega_final<<<S, TPB, 0, stream>>>(i_f, Ztp_f, Remap_f, Wfin, out, D);
}
